// GCN_70909910057300
// MI455X (gfx1250) — matrix operand order repaired and checked by reading, not compiled, not run on hardware
//
#include <hip/hip_runtime.h>
#include <hip/hip_bf16.h>

#define N_NODES 50000   // == 3125 * 16  -> M tiles divide exactly, no row guards
#define N_EDGES 600000

typedef __attribute__((ext_vector_type(16))) __bf16        v16bf;
typedef __attribute__((ext_vector_type(8)))  float         v8f;
typedef __attribute__((ext_vector_type(8)))  unsigned int  v8u;   // 32B raw = one B-fragment per lane

// ---------------------------------------------------------------------------
// Utility: zero-fill a float buffer
// ---------------------------------------------------------------------------
__global__ void fill0_kernel(float* __restrict__ p, long long n) {
    long long i = (long long)blockIdx.x * blockDim.x + threadIdx.x;
    if (i < n) p[i] = 0.0f;
}

// ---------------------------------------------------------------------------
// In-degree count (float atomics), then invert in place: cnt -> 1/max(cnt,1)
// ---------------------------------------------------------------------------
__global__ void deg_count_kernel(const int* __restrict__ dst, float* __restrict__ cnt, int E) {
    int i = blockIdx.x * blockDim.x + threadIdx.x;
    if (i < E) atomicAdd(&cnt[dst[i]], 1.0f);
}

__global__ void deg_inv_kernel(float* __restrict__ cnt, int N) {
    int i = blockIdx.x * blockDim.x + threadIdx.x;
    if (i < N) cnt[i] = 1.0f / fmaxf(cnt[i], 1.0f);
}

// ---------------------------------------------------------------------------
// Pad layer-1 features: x [N,2] -> xpad [N,32] (cols 2..31 = 0)
// ---------------------------------------------------------------------------
__global__ void pad_x_kernel(const float* __restrict__ x, float* __restrict__ xpad, int total) {
    int t = blockIdx.x * blockDim.x + threadIdx.x;
    if (t >= total) return;
    int node = t >> 5;
    int k    = t & 31;
    xpad[t] = (k < 2) ? x[node * 2 + k] : 0.0f;
}

// ---------------------------------------------------------------------------
// Repack a weight matrix W [din x dout] (f32, row-major) into bf16 WMMA
// B-fragment order:  out[((tile*nchunks + c)*32 + lane)*16 + i]
//   k = c*32 + (lane>>4)*8 + (i&7) + (i>>3)*16   (zero-padded for k >= din_real)
//   n = tile*16 + (lane&15)
// Each lane's 16 bf16 values become one contiguous 32-byte load in the GEMM.
// ---------------------------------------------------------------------------
__global__ void pack_w_kernel(const float* __restrict__ W, __bf16* __restrict__ out,
                              int din_real, int dout, int nchunks, int total) {
    int t = blockIdx.x * blockDim.x + threadIdx.x;
    if (t >= total) return;
    int i    = t & 15;
    int lane = (t >> 4) & 31;
    int rest = t >> 9;                 // tile*nchunks + c
    int c    = rest % nchunks;
    int tile = rest / nchunks;
    int k = c * 32 + (lane >> 4) * 8 + (i & 7) + (i >> 3) * 16;
    int n = tile * 16 + (lane & 15);
    float v = (k < din_real) ? W[(size_t)k * dout + n] : 0.0f;
    out[t] = (__bf16)v;
}

// ---------------------------------------------------------------------------
// Edge scatter-add, float4 granularity: agg[dst[e]][4q..] += x[src[e]][4q..]
// Gather + atomic tables are <=25.6MB: L2-resident on MI455X (192MB L2).
// ---------------------------------------------------------------------------
__global__ void scatter_add4_kernel(const int* __restrict__ src, const int* __restrict__ dst,
                                    const float* __restrict__ x, float* __restrict__ agg,
                                    int log2din4, long long total) {
    long long idx = (long long)blockIdx.x * blockDim.x + threadIdx.x;
    if (idx >= total) return;
    int e  = (int)(idx >> log2din4);
    int q  = (int)(idx & ((1LL << log2din4) - 1));
    int d4 = 1 << log2din4;                       // din/4
    const float4 v = ((const float4*)x)[(size_t)src[e] * d4 + q];
    float* a = agg + (((size_t)dst[e] * d4 + q) << 2);
    atomicAdd(a + 0, v.x);
    atomicAdd(a + 1, v.y);
    atomicAdd(a + 2, v.z);
    atomicAdd(a + 3, v.w);
}

// ---------------------------------------------------------------------------
// Fused SAGE layer:  hout = relu( (agg*deg_inv) @ Wl + hin @ Wr + b )
// One wave computes a 16x16 output tile with v_wmma_f32_16x16x32_bf16.
// DIN is a multiple of 32; M = 50000 divides into 16-row tiles exactly;
// block has exactly dout/16 waves -> NO guards, EXEC all-1s throughout.
// A fragment: 4 contiguous global_load_b128 per operand.
// B fragment: one 32B vector load per matrix from the packed bf16 table.
// ---------------------------------------------------------------------------
template <int DIN>
__global__ void sage_gemm_kernel(const float* __restrict__ hin,
                                 const float* __restrict__ agg,
                                 const float* __restrict__ deg_inv,
                                 const __bf16* __restrict__ wlpack,
                                 const __bf16* __restrict__ wrpack,
                                 const float* __restrict__ bias,
                                 float* __restrict__ hout, int dout) {
    constexpr int NCH = DIN / 32;
    const int lane = threadIdx.x & 31;
    const int wave = threadIdx.x >> 5;
    const int half = lane >> 4;
    const int l15  = lane & 15;
    const int row0 = blockIdx.x * 16;

    const int   m    = row0 + l15;            // A-fragment row for this lane
    const float dinv = deg_inv[m];
    const int   n    = wave * 16 + l15;       // B/C/D column for this lane

    const float* hrow = hin + (size_t)m * DIN;
    const float* arow = agg + (size_t)m * DIN;
    const v8u* wlp = (const v8u*)wlpack + (size_t)wave * NCH * 32;
    const v8u* wrp = (const v8u*)wrpack + (size_t)wave * NCH * 32;

    v8f c = {};
    #pragma unroll
    for (int ch = 0; ch < NCH; ++ch) {
        const int kb = ch * 32 + half * 8;
        // ---- A (self) : 16 contiguous bf16 per lane from 2x(2xfloat4)
        float4 s0 = *(const float4*)(hrow + kb);
        float4 s1 = *(const float4*)(hrow + kb + 4);
        float4 s2 = *(const float4*)(hrow + kb + 16);
        float4 s3 = *(const float4*)(hrow + kb + 20);
        // ---- A (mean) = agg * deg_inv
        float4 m0 = *(const float4*)(arow + kb);
        float4 m1 = *(const float4*)(arow + kb + 4);
        float4 m2 = *(const float4*)(arow + kb + 16);
        float4 m3 = *(const float4*)(arow + kb + 20);

        float sv[16] = {s0.x, s0.y, s0.z, s0.w, s1.x, s1.y, s1.z, s1.w,
                        s2.x, s2.y, s2.z, s2.w, s3.x, s3.y, s3.z, s3.w};
        float mv[16] = {m0.x, m0.y, m0.z, m0.w, m1.x, m1.y, m1.z, m1.w,
                        m2.x, m2.y, m2.z, m2.w, m3.x, m3.y, m3.z, m3.w};

        v16bf a_self, a_mean;
        #pragma unroll
        for (int i = 0; i < 16; ++i) {
            a_self[i] = (__bf16)sv[i];
            a_mean[i] = (__bf16)(mv[i] * dinv);
        }

        // ---- B fragments: one 32B load each (packed bf16, L2-resident)
        v16bf b_l = __builtin_bit_cast(v16bf, wlp[ch * 32 + lane]);
        v16bf b_r = __builtin_bit_cast(v16bf, wrp[ch * 32 + lane]);

        c = __builtin_amdgcn_wmma_f32_16x16x32_bf16(false, a_mean, false, b_l,
                                                    (short)0, c, false, false);
        c = __builtin_amdgcn_wmma_f32_16x16x32_bf16(false, a_self, false, b_r,
                                                    (short)0, c, false, false);
    }

    // ---- epilogue: +bias, ReLU, store.  D layout: M = j + 8*half, N = n
    const float bv = bias[n];
    #pragma unroll
    for (int j = 0; j < 8; ++j) {
        int mrow = row0 + half * 8 + j;
        hout[(size_t)mrow * dout + n] = fmaxf(c[j] + bv, 0.0f);
    }
}

// ---------------------------------------------------------------------------
// Final projection: out[i] = dot(h[i], Wlin) + blin   (128 -> 1)
// ---------------------------------------------------------------------------
__global__ void final_linear_kernel(const float* __restrict__ h,
                                    const float* __restrict__ Wlin,
                                    const float* __restrict__ blin,
                                    float* __restrict__ out, int N) {
    int i = blockIdx.x * blockDim.x + threadIdx.x;
    if (i >= N) return;
    const float4* row = (const float4*)(h + (size_t)i * 128);
    const float4* w   = (const float4*)Wlin;
    float s = blin[0];
    #pragma unroll
    for (int k = 0; k < 32; ++k) {
        float4 a = row[k], b = w[k];
        s += a.x * b.x + a.y * b.y + a.z * b.z + a.w * b.w;
    }
    out[i] = s;
}

// ---------------------------------------------------------------------------
// Host-side drivers
// ---------------------------------------------------------------------------
static void run_sage_layer(const float* hin, float* agg, float* hout,
                           const float* deg_inv, const int* src, const int* dst,
                           const __bf16* wlp, const __bf16* wrp, const float* b,
                           int din_p, int dout, hipStream_t stream) {
    long long aggN = (long long)N_NODES * din_p;
    fill0_kernel<<<(int)((aggN + 255) / 256), 256, 0, stream>>>(agg, aggN);

    int lg4 = (din_p == 32) ? 3 : 5;                    // log2(din_p/4)
    long long total = (long long)N_EDGES << lg4;
    scatter_add4_kernel<<<(int)((total + 255) / 256), 256, 0, stream>>>(
        src, dst, hin, agg, lg4, total);

    dim3 grid(N_NODES / 16);                            // 3125, exact
    dim3 block((dout / 16) * 32);                       // 64 or 256 threads
    if (din_p == 32)
        sage_gemm_kernel<32><<<grid, block, 0, stream>>>(hin, agg, deg_inv, wlp, wrp, b, hout, dout);
    else
        sage_gemm_kernel<128><<<grid, block, 0, stream>>>(hin, agg, deg_inv, wlp, wrp, b, hout, dout);
}

static void pack_w(const float* W, __bf16* out, int din_real, int din_p, int dout,
                   hipStream_t stream) {
    int nchunks = din_p / 32;
    int total = (dout / 16) * nchunks * 512;
    pack_w_kernel<<<(total + 255) / 256, 256, 0, stream>>>(W, out, din_real, dout, nchunks, total);
}

extern "C" void kernel_launch(void* const* d_in, const int* in_sizes, int n_in,
                              void* d_out, int out_size, void* d_ws, size_t ws_size,
                              hipStream_t stream) {
    (void)in_sizes; (void)n_in; (void)out_size; (void)ws_size;

    const float* x   = (const float*)d_in[0];
    const int*   ei  = (const int*)d_in[1];
    const int*   src = ei;             // edge_index row 0
    const int*   dst = ei + N_EDGES;   // edge_index row 1

    const float* W[5][3];   // Wl, b, Wr per layer
    for (int i = 0; i < 5; ++i) {
        W[i][0] = (const float*)d_in[2 + 3 * i];
        W[i][1] = (const float*)d_in[3 + 3 * i];
        W[i][2] = (const float*)d_in[4 + 3 * i];
    }
    const float* Wlin = (const float*)d_in[17];
    const float* blin = (const float*)d_in[18];

    // workspace layout (256B-aligned slabs)
    char* ws = (char*)d_ws;
    size_t off = 0;
    auto grab = [&](size_t bytes) {
        char* p = ws + off;
        off += (bytes + 255) & ~(size_t)255;
        return p;
    };
    float* deg  = (float*)grab((size_t)N_NODES * sizeof(float));            // 0.2 MB
    float* xpad = (float*)grab((size_t)N_NODES * 32 * sizeof(float));       // 6.4 MB
    float* agg  = (float*)grab((size_t)N_NODES * 128 * sizeof(float));      // 25.6 MB
    float* hA   = (float*)grab((size_t)N_NODES * 128 * sizeof(float));      // 25.6 MB
    float* hB   = (float*)grab((size_t)N_NODES * 128 * sizeof(float));      // 25.6 MB

    // packed bf16 weights (per layer): [dout/16][din_p/32][32 lanes][16]
    const int din_real[5] = {2, 32, 128, 128, 128};
    const int din_p[5]    = {32, 32, 128, 128, 128};
    const int dout_[5]    = {32, 128, 128, 128, 128};
    __bf16* wlp[5]; __bf16* wrp[5];
    for (int i = 0; i < 5; ++i) {
        size_t nelem = (size_t)(dout_[i] / 16) * (din_p[i] / 32) * 512;
        wlp[i] = (__bf16*)grab(nelem * sizeof(__bf16));
        wrp[i] = (__bf16*)grab(nelem * sizeof(__bf16));
    }

    // ---- one-shot preprocessing (cheap; part of every deterministic replay)
    for (int i = 0; i < 5; ++i) {
        pack_w(W[i][0], wlp[i], din_real[i], din_p[i], dout_[i], stream);
        pack_w(W[i][2], wrp[i], din_real[i], din_p[i], dout_[i], stream);
    }
    pad_x_kernel<<<(N_NODES * 32 + 255) / 256, 256, 0, stream>>>(x, xpad, N_NODES * 32);

    // ---- degree -> deg_inv (in place)
    fill0_kernel<<<(N_NODES + 255) / 256, 256, 0, stream>>>(deg, N_NODES);
    deg_count_kernel<<<(N_EDGES + 255) / 256, 256, 0, stream>>>(dst, deg, N_EDGES);
    deg_inv_kernel<<<(N_NODES + 255) / 256, 256, 0, stream>>>(deg, N_NODES);

    // ---- 5 SAGE layers (ping-pong hA/hB)
    run_sage_layer(xpad, agg, hA, deg, src, dst, wlp[0], wrp[0], W[0][1],  32,  32, stream);
    run_sage_layer(hA,   agg, hB, deg, src, dst, wlp[1], wrp[1], W[1][1],  32, 128, stream);
    run_sage_layer(hB,   agg, hA, deg, src, dst, wlp[2], wrp[2], W[2][1], 128, 128, stream);
    run_sage_layer(hA,   agg, hB, deg, src, dst, wlp[3], wrp[3], W[3][1], 128, 128, stream);
    run_sage_layer(hB,   agg, hA, deg, src, dst, wlp[4], wrp[4], W[4][1], 128, 128, stream);

    // ---- final 128->1 projection
    final_linear_kernel<<<(N_NODES + 255) / 256, 256, 0, stream>>>(
        hA, Wlin, blin, (float*)d_out, N_NODES);
}
